// MLP_75840532512772
// MI455X (gfx1250) — matrix operand order repaired and checked by reading, not compiled, not run on hardware
//
#include <hip/hip_runtime.h>
#include <hip/hip_bf16.h>

typedef __attribute__((ext_vector_type(16))) _Float16 v16h;
typedef __attribute__((ext_vector_type(8)))  _Float16 v8h;
typedef __attribute__((ext_vector_type(8)))  float    v8f;

// ---------------------------------------------------------------------------
// Weight packing: W[K][N] f32 (row-major, fan_in x fan_out) -> f16 WMMA
// B-fragments. Fragment f = nt*numKT + kt holds a 32(K) x 16(N) tile:
//   lane l<16 : K = kt*32 +      h, N = nt*16 + l        (h = half 0..15)
//   lane l>=16: K = kt*32 + 16 + h, N = nt*16 + (l-16)
// Packed element index = f*512 + lane*16 + h  (each lane reads 32B contiguous)
// ---------------------------------------------------------------------------
__global__ __launch_bounds__(256) void pack_weights_kernel(
    const float* __restrict__ W, _Float16* __restrict__ Wp,
    int Kreal, int Kpad, int N) {
  int numKT = Kpad >> 5;
  int total = (N >> 4) * numKT * 512;
  for (int idx = blockIdx.x * blockDim.x + threadIdx.x; idx < total;
       idx += gridDim.x * blockDim.x) {
    int h  = idx & 15;
    int l  = (idx >> 4) & 31;
    int f  = idx >> 9;
    int kt = f % numKT;
    int nt = f / numKT;
    int K  = kt * 32 + ((l >> 4) << 4) + h;
    int n  = nt * 16 + (l & 15);
    float v = (K < Kreal) ? W[(size_t)K * N + n] : 0.0f;
    Wp[idx] = (_Float16)v;
  }
}

// A-fragment (16x32 f16) from row-major f16 LDS tile, stride Kpad halves.
// lane l<16 : M=l,    halves 0..15 = K kt*32+0..15
// lane l>=16: M=l-16, halves 0..15 = K kt*32+16..31
__device__ __forceinline__ v16h load_a_frag(const _Float16* lds_in, int Kpad,
                                            int kt, int lane) {
  int row = lane & 15;
  int kb  = kt * 32 + ((lane >> 4) << 4);
  const _Float16* p = lds_in + row * Kpad + kb;
  v8h lo = *(const v8h*)p;
  v8h hi = *(const v8h*)(p + 8);
  return __builtin_shufflevector(lo, hi, 0, 1, 2, 3, 4, 5, 6, 7,
                                         8, 9, 10, 11, 12, 13, 14, 15);
}

// 64-row GEMM slice: [64,Kpad](LDS f16) x [Kpad, 16*ntCount](packed f16)
// + bias, optional ReLU, f16 result to LDS. Four 16-row M-tiles share every
// B fragment (one global 32B/lane load feeds four WMMAs -> 4x B reuse).
// Each wave handles N-tiles [ntBase, ntBase+ntCount).
__device__ __forceinline__ void gemm_layer(
    const _Float16* lds_in, _Float16* lds_out,
    const _Float16* __restrict__ Wp, const float* __restrict__ bias,
    int Kpad, int outStride, bool do_relu, int lane,
    int ntBase, int ntCount) {
  const int numKT = Kpad >> 5;
  const int rbase = (lane >> 4) << 3;   // C rows: lanes 0-15 -> M=r, 16-31 -> M=r+8
  const int nsub  = lane & 15;
  for (int nt = ntBase; nt < ntBase + ntCount; ++nt) {
    v8f acc0 = {};
    v8f acc1 = {};
    v8f acc2 = {};
    v8f acc3 = {};
    const _Float16* wp = Wp + ((size_t)nt * numKT) * 512 + lane * 16;
    for (int kt = 0; kt < numKT; ++kt) {
      v16h b  = *(const v16h*)wp;   // contiguous 32B per lane (L2-resident)
      wp += 512;
      v16h a0 = load_a_frag(lds_in,             Kpad, kt, lane);
      v16h a1 = load_a_frag(lds_in + 16 * Kpad, Kpad, kt, lane);
      v16h a2 = load_a_frag(lds_in + 32 * Kpad, Kpad, kt, lane);
      v16h a3 = load_a_frag(lds_in + 48 * Kpad, Kpad, kt, lane);
      acc0 = __builtin_amdgcn_wmma_f32_16x16x32_f16(
          false, a0, false, b, (short)0, acc0, false, false);
      acc1 = __builtin_amdgcn_wmma_f32_16x16x32_f16(
          false, a1, false, b, (short)0, acc1, false, false);
      acc2 = __builtin_amdgcn_wmma_f32_16x16x32_f16(
          false, a2, false, b, (short)0, acc2, false, false);
      acc3 = __builtin_amdgcn_wmma_f32_16x16x32_f16(
          false, a3, false, b, (short)0, acc3, false, false);
    }
    int col  = nt * 16 + nsub;
    float bv = bias[col];
#pragma unroll
    for (int r = 0; r < 8; ++r) {
      float v0 = acc0[r] + bv;
      float v1 = acc1[r] + bv;
      float v2 = acc2[r] + bv;
      float v3 = acc3[r] + bv;
      if (do_relu) {
        v0 = v0 > 0.0f ? v0 : 0.0f;
        v1 = v1 > 0.0f ? v1 : 0.0f;
        v2 = v2 > 0.0f ? v2 : 0.0f;
        v3 = v3 > 0.0f ? v3 : 0.0f;
      }
      lds_out[(rbase + r) * outStride + col]        = (_Float16)v0;
      lds_out[(16 + rbase + r) * outStride + col]   = (_Float16)v1;
      lds_out[(32 + rbase + r) * outStride + col]   = (_Float16)v2;
      lds_out[(48 + rbase + r) * outStride + col]   = (_Float16)v3;
    }
  }
}

// ---------------------------------------------------------------------------
// Fully fused MLP: 4 waves per block share a 64-row tile; activations stay in
// LDS (116KB/block -> 2 blocks/WGP, 8 waves/WGP = 2/SIMD). Waves split the
// N-tiles of every layer 4 ways, so each B fragment is fetched once per 64
// rows (4x M reuse per wave on top of the 4-way N split). x is read once,
// non-temporal, keeping packed weights resident in the 192MB L2.
// ---------------------------------------------------------------------------
__global__ __launch_bounds__(128) void mlp_fused_kernel(
    const float* __restrict__ x,
    const _Float16* __restrict__ W1p, const float* __restrict__ b1,
    const _Float16* __restrict__ W2p, const float* __restrict__ b2,
    const _Float16* __restrict__ W3p, const float* __restrict__ b3,
    const float* __restrict__ Ws,  const float* __restrict__ bs,
    const _Float16* __restrict__ Wc1p, const float* __restrict__ bc1,
    const _Float16* __restrict__ Wc2p, const float* __restrict__ bc2,
    const float* __restrict__ Wc3, const float* __restrict__ bc3,
    float* __restrict__ out) {
  __shared__ alignas(16) _Float16 buf0[64 * 384];  // x / h2 / c1    (48KB)
  __shared__ alignas(16) _Float16 buf1[64 * 512];  // h1 / comb / c2 (64KB)
  __shared__ _Float16 statp[64][32];               // stats @ Ws + bs (4KB)

  const int w    = threadIdx.x >> 5;   // wave id (0..3)
  const int lane = threadIdx.x & 31;
  const int r0   = blockIdx.x * 64;

  // ---- Stage 1: x tile f32->f16 into LDS (K padded 365->384) + fused
  //      row moments + 6->32 stat projection. Wave w owns rows w*16..w*16+15.
  const float invD = 1.0f / 365.0f;
  for (int i = 0; i < 16; ++i) {
    const int m = w * 16 + i;
    const float* xr = x + (size_t)(r0 + m) * 365;
    float s1 = 0.f, s2 = 0.f, s3 = 0.f, s4 = 0.f;
    float mn = 3.4e38f, mx = -3.4e38f;
    for (int c = lane; c < 384; c += 32) {
      float v = (c < 365) ? __builtin_nontemporal_load(&xr[c]) : 0.0f;
      buf0[m * 384 + c] = (_Float16)v;
      if (c < 365) {
        s1 += v; float v2 = v * v; s2 += v2; s3 += v2 * v; s4 += v2 * v2;
        mn = fminf(mn, v); mx = fmaxf(mx, v);
      }
    }
#pragma unroll
    for (int off = 16; off >= 1; off >>= 1) {
      s1 += __shfl_xor(s1, off);
      s2 += __shfl_xor(s2, off);
      s3 += __shfl_xor(s3, off);
      s4 += __shfl_xor(s4, off);
      mn = fminf(mn, __shfl_xor(mn, off));
      mx = fmaxf(mx, __shfl_xor(mx, off));
    }
    float mu  = s1 * invD;
    float e2  = s2 * invD, e3 = s3 * invD, e4 = s4 * invD;
    float var1 = (s2 - mu * s1) / 364.0f;     // unbiased (ddof=1)
    float sig  = sqrtf(var1);
    float m3   = e3 - 3.0f * mu * e2 + 2.0f * mu * mu * mu;
    float m4   = e4 - 4.0f * mu * e3 + 6.0f * mu * mu * e2
                    - 3.0f * mu * mu * mu * mu;
    float s3p  = sig * sig * sig;
    float skew = m3 / (s3p + 1e-8f);
    float kurt = m4 / (s3p * sig + 1e-8f);
    // project [mu, sig, mn, mx, skew, kurt] by Ws[6,32]; this lane owns n=lane
    statp[m][lane] = (_Float16)(bs[lane]
        + mu   * Ws[0 * 32 + lane] + sig  * Ws[1 * 32 + lane]
        + mn   * Ws[2 * 32 + lane] + mx   * Ws[3 * 32 + lane]
        + skew * Ws[4 * 32 + lane] + kurt * Ws[5 * 32 + lane]);
  }
  __syncthreads();

  // L1: [64,384] @ [384,512] -> h1 (buf1); wave w owns N-tiles w*8..+8
  gemm_layer(buf0, buf1, W1p, b1, 384, 512, true, lane, w * 8, 8);
  __syncthreads();
  // L2: [64,512] @ [512,256] -> h2 (buf0)
  gemm_layer(buf1, buf0, W2p, b2, 512, 256, true, lane, w * 4, 4);
  __syncthreads();
  // L3: [64,256] @ [256,128] -> comb cols 0..127 (buf1, stride 160)
  gemm_layer(buf0, buf1, W3p, b3, 256, 160, true, lane, w * 2, 2);
  // comb cols 128..159 = stat branch (no ReLU, matching reference)
  for (int i = 0; i < 16; ++i) {
    const int m = w * 16 + i;
    buf1[m * 160 + 128 + lane] = statp[m][lane];
  }
  __syncthreads();
  // Lc1: [64,160] @ [160,64] -> c1 (buf0, stride 64)
  gemm_layer(buf1, buf0, Wc1p, bc1, 160, 64, true, lane, w, 1);
  __syncthreads();
  // Lc2: [64,64] @ [64,32] -> c2 (buf1 + 16384, stride 32); 2 tiles -> waves 0,1
  gemm_layer(buf0, buf1 + 16384, Wc2p, bc2, 64, 32, true, lane,
             w, (w < 2) ? 1 : 0);
  __syncthreads();

  // Head: [64,32] @ [32,1] + sigmoid (VALU; trivial FLOPs). Thread = row.
  if (threadIdx.x < 64) {
    const int m = threadIdx.x;
    const _Float16* c2 = buf1 + 16384 + m * 32;
    float acc = bc3[0];
#pragma unroll
    for (int j = 0; j < 32; ++j) acc += (float)c2[j] * Wc3[j];
    float r = 1.0f / (1.0f + expf(-acc));
    __builtin_nontemporal_store(r, &out[r0 + m]);
  }
}

extern "C" void kernel_launch(void* const* d_in, const int* in_sizes, int n_in,
                              void* d_out, int out_size, void* d_ws,
                              size_t ws_size, hipStream_t stream) {
  (void)in_sizes; (void)n_in; (void)out_size; (void)ws_size;
  const float* x   = (const float*)d_in[0];
  const float* W1  = (const float*)d_in[1];
  const float* b1  = (const float*)d_in[2];
  const float* W2  = (const float*)d_in[3];
  const float* b2  = (const float*)d_in[4];
  const float* W3  = (const float*)d_in[5];
  const float* b3  = (const float*)d_in[6];
  const float* Ws  = (const float*)d_in[7];
  const float* bs  = (const float*)d_in[8];
  const float* Wc1 = (const float*)d_in[9];
  const float* bc1 = (const float*)d_in[10];
  const float* Wc2 = (const float*)d_in[11];
  const float* bc2 = (const float*)d_in[12];
  const float* Wc3 = (const float*)d_in[13];
  const float* bc3 = (const float*)d_in[14];

  // Workspace layout: packed f16 weights (repacked every call; deterministic).
  _Float16* ws  = (_Float16*)d_ws;
  _Float16* W1p  = ws;              // 12*32*512 = 196608 halves
  _Float16* W2p  = W1p + 196608;    // 16*16*512 = 131072
  _Float16* W3p  = W2p + 131072;    //  8* 8*512 =  32768
  _Float16* Wc1p = W3p + 32768;     //  5* 4*512 =  10240
  _Float16* Wc2p = Wc1p + 10240;    //  2* 2*512 =   2048   (~729 KB total)

  pack_weights_kernel<<<384, 256, 0, stream>>>(W1,  W1p,  365, 384, 512);
  pack_weights_kernel<<<256, 256, 0, stream>>>(W2,  W2p,  512, 512, 256);
  pack_weights_kernel<<<64,  256, 0, stream>>>(W3,  W3p,  256, 256, 128);
  pack_weights_kernel<<<20,  256, 0, stream>>>(Wc1, Wc1p, 160, 160, 64);
  pack_weights_kernel<<<4,   256, 0, stream>>>(Wc2, Wc2p, 64,  64,  32);

  // 65536 rows / 64 rows-per-block = 1024 blocks of 4 waves
  mlp_fused_kernel<<<1024, 128, 0, stream>>>(
      x, W1p, b1, W2p, b2, W3p, b3, Ws, bs, Wc1p, bc1, Wc2p, bc2, Wc3, bc3,
      (float*)d_out);
}
